// NeuralODE_15556371546632
// MI455X (gfx1250) — compile-verified, matrix operand order checked by reading, not run on hardware
//
#include <hip/hip_runtime.h>

typedef __attribute__((ext_vector_type(16))) _Float16 v16h;
typedef __attribute__((ext_vector_type(8)))  float    v8f;

#define M_TRAJ 4096
#define N_STEPS 200
#define HID 128
#define WAVES_PER_BLOCK 4
#define ROWS_PER_WAVE 16
#define ROWS_PER_BLOCK (WAVES_PER_BLOCK * ROWS_PER_WAVE)  // 64

// One wave owns 16 trajectories (one WMMA M-tile) for the full 199-step RK4
// integration. All weights live in LDS; W2 is pre-packed into the CDNA5
// 16-bit B-matrix lane layout so each wave can feed v_wmma_f32_16x16x32_f16
// straight from ds_load_b128.
__global__ __launch_bounds__(ROWS_PER_BLOCK * 2, 1) void node_rk4_kernel(
    const float* __restrict__ x0, const float* __restrict__ t,
    const float* __restrict__ W1, const float* __restrict__ b1,
    const float* __restrict__ W2, const float* __restrict__ b2,
    const float* __restrict__ W3, const float* __restrict__ b3,
    float* __restrict__ out)
{
    // W2 packed as B tiles: [kt][nt][lane][elem], tile = K32 x N16
    __shared__ __attribute__((aligned(32))) _Float16 sWb[4][8][32][16]; // 32 KB
    __shared__ float sW1[2][HID];
    __shared__ float sB1[HID];
    __shared__ float sB2[HID];
    __shared__ float sW3[HID][2];
    __shared__ float sX[WAVES_PER_BLOCK][ROWS_PER_WAVE][2];  // per-wave transpose scratch

    const int tid  = threadIdx.x;
    const int lane = tid & 31;
    const int wave = tid >> 5;
    const int hi   = (lane >> 4) & 1;   // 0: lanes 0-15, 1: lanes 16-31
    const int l15  = lane & 15;

    // ---- pack weights into LDS (once per block) ----
    // B tile (kt,nt): lane holds col n = 16*nt + (lane&15);
    // element i holds K = 32*kt + (i<8 ? 8*hi + i : 16 + 8*hi + (i-8))
    for (int idx = tid; idx < 4 * 8 * 32 * 16; idx += blockDim.x) {
        int i   = idx & 15;
        int ln  = (idx >> 4) & 31;
        int nt  = (idx >> 9) & 7;
        int kt  = (idx >> 12) & 3;
        int lhi = (ln >> 4) & 1;
        int kk  = (i < 8) ? (8 * lhi + i) : (16 + 8 * lhi + (i - 8));
        int k   = 32 * kt + kk;
        int n   = 16 * nt + (ln & 15);
        sWb[kt][nt][ln][i] = (_Float16)W2[k * HID + n];
    }
    for (int idx = tid; idx < 2 * HID; idx += blockDim.x) sW1[idx >> 7][idx & 127] = W1[idx];
    for (int idx = tid; idx < HID; idx += blockDim.x) { sB1[idx] = b1[idx]; sB2[idx] = b2[idx]; }
    for (int idx = tid; idx < 2 * HID; idx += blockDim.x) sW3[idx >> 1][idx & 1] = W3[idx];
    __syncthreads();

    // ---- per-lane loop-invariant registers ----
    float b2r[8], w3r0[8], w3r1[8];
#pragma unroll
    for (int nt = 0; nt < 8; ++nt) {
        b2r[nt]  = sB2[16 * nt + l15];        // lane == column in C/D layout
        w3r0[nt] = sW3[16 * nt + l15][0];
        w3r1[nt] = sW3[16 * nt + l15][1];
    }
    const float b30 = b3[0], b31 = b3[1];

    const int row0 = blockIdx.x * ROWS_PER_BLOCK + wave * ROWS_PER_WAVE;
    const int m    = row0 + l15;               // this lane's trajectory row
    float xr0 = x0[m * 2 + 0];
    float xr1 = x0[m * 2 + 1];
    if (hi == 0) { out[m * 2 + 0] = xr0; out[m * 2 + 1] = xr1; }  // traj[0] = x0

    // f(x): 3-layer MLP for this wave's 16 rows. Input/output in row-per-lane
    // layout (lane holds row lane&15, lo/hi halves replicated).
    auto evalf = [&](float in0, float in1, float& k0, float& k1) {
        // layer 1 -> A tiles, built directly in 16x32 f16 A-matrix lane layout
        v16h a[4];
#pragma unroll
        for (int c = 0; c < 4; ++c) {
#pragma unroll
            for (int i = 0; i < 16; ++i) {
                int nn  = 32 * c + ((i < 8) ? (8 * hi + i) : (16 + 8 * hi + (i - 8)));
                float h = fmaf(in0, sW1[0][nn], fmaf(in1, sW1[1][nn], sB1[nn]));
                a[c][i] = (_Float16)fmaxf(h, 0.0f);
            }
        }
        // layer 2 (WMMA, fp32 accumulate, bias-seeded) fused with layer-3 dot
        float y0[8], y1[8];
#pragma unroll
        for (int v = 0; v < 8; ++v) { y0[v] = 0.0f; y1[v] = 0.0f; }
#pragma unroll
        for (int nt = 0; nt < 8; ++nt) {
            v8f acc;
#pragma unroll
            for (int v = 0; v < 8; ++v) acc[v] = b2r[nt];
#pragma unroll
            for (int kt = 0; kt < 4; ++kt) {
                v16h bt = *(const v16h*)(&sWb[kt][nt][lane][0]);
                acc = __builtin_amdgcn_wmma_f32_16x16x32_f16(
                    false, a[kt], false, bt, (short)0, acc, false, false);
            }
#pragma unroll
            for (int v = 0; v < 8; ++v) {   // relu + W3 partial dot (lane's column)
                float h2 = fmaxf(acc[v], 0.0f);
                y0[v] = fmaf(h2, w3r0[nt], y0[v]);
                y1[v] = fmaf(h2, w3r1[nt], y1[v]);
            }
        }
        // reduce across the 16 column-lanes of each half (rows stay put)
#pragma unroll
        for (int mask = 1; mask < 16; mask <<= 1) {
#pragma unroll
            for (int v = 0; v < 8; ++v) {
                y0[v] += __shfl_xor(y0[v], mask, 32);
                y1[v] += __shfl_xor(y1[v], mask, 32);
            }
        }
        // lanes 0-15 now hold rows 0-7, lanes 16-31 rows 8-15 -> transpose via
        // per-wave LDS back to row-per-lane layout (intra-wave, DS in-order)
        if (l15 == 0) {
#pragma unroll
            for (int v = 0; v < 8; ++v) {
                sX[wave][8 * hi + v][0] = y0[v] + b30;
                sX[wave][8 * hi + v][1] = y1[v] + b31;
            }
        }
        asm volatile("s_wait_dscnt 0x0" ::: "memory");
        k0 = sX[wave][l15][0];
        k1 = sX[wave][l15][1];
    };

#pragma unroll 1
    for (int step = 0; step < N_STEPS - 1; ++step) {
        float h = t[step + 1] - t[step];
        float k10, k11, k20, k21, k30, k31, k40, k41;
        evalf(xr0, xr1, k10, k11);
        evalf(fmaf(0.5f * h, k10, xr0), fmaf(0.5f * h, k11, xr1), k20, k21);
        evalf(fmaf(0.5f * h, k20, xr0), fmaf(0.5f * h, k21, xr1), k30, k31);
        evalf(fmaf(h, k30, xr0), fmaf(h, k31, xr1), k40, k41);
        float s = h * (1.0f / 6.0f);
        xr0 += s * (k10 + 2.0f * k20 + 2.0f * k30 + k40);
        xr1 += s * (k11 + 2.0f * k21 + 2.0f * k31 + k41);
        if (hi == 0) {
            float* o = out + (size_t)(step + 1) * (M_TRAJ * 2);
            o[m * 2 + 0] = xr0;
            o[m * 2 + 1] = xr1;
        }
    }
}

extern "C" void kernel_launch(void* const* d_in, const int* in_sizes, int n_in,
                              void* d_out, int out_size, void* d_ws, size_t ws_size,
                              hipStream_t stream) {
    const float* x0 = (const float*)d_in[0];
    const float* t  = (const float*)d_in[1];
    const float* W1 = (const float*)d_in[2];
    const float* b1 = (const float*)d_in[3];
    const float* W2 = (const float*)d_in[4];
    const float* b2 = (const float*)d_in[5];
    const float* W3 = (const float*)d_in[6];
    const float* b3 = (const float*)d_in[7];
    float* out = (float*)d_out;

    dim3 grid(M_TRAJ / ROWS_PER_BLOCK);   // 64 blocks
    dim3 block(ROWS_PER_BLOCK * 2);       // 128 threads = 4 waves
    hipLaunchKernelGGL(node_rk4_kernel, grid, block, 0, stream,
                       x0, t, W1, b1, W2, b2, W3, b3, out);
}
